// RNN_CRF_51384988729505
// MI455X (gfx1250) — matrix operand order repaired and checked by reading, not compiled, not run on hardware
//
#include <hip/hip_runtime.h>
#include <hip/hip_bf16.h>
#include <stdint.h>

// ---------------------------------------------------------------------------
// BiLSTM + CRF Viterbi for MI455X (gfx1250, wave32, WMMA f32_16x16x32_f16,
// TDM tensor_load_to_lds for A-panel staging)
// ---------------------------------------------------------------------------
#define V_   50000
#define E_   256
#define H_   512
#define G4H  2048          // 4*H
#define K_   32
#define B_   32
#define T_   256
#define TB   (T_*B_)       // 8192
#define START_ 30
#define STOP_  31
#define NEG_  (-10000.0f)

typedef _Float16 v16h __attribute__((ext_vector_type(16)));
typedef float    v8f  __attribute__((ext_vector_type(8)));
typedef unsigned int u32x4 __attribute__((ext_vector_type(4)));
typedef int          i32x4 __attribute__((ext_vector_type(4)));
typedef int          i32x8 __attribute__((ext_vector_type(8)));

union Frag16 {            // 16 f16 = one WMMA A/B operand (32 bytes)
    v16h  v;
    float4 f4[2];
};
union Half8 {             // 8 f16 = one C-fragment row slice (16 bytes)
    _Float16 h[8];
    float4   f4;
};

__device__ __forceinline__ v8f wmma_f16(v16h a, v16h b, v8f c) {
    return __builtin_amdgcn_wmma_f32_16x16x32_f16(false, a, false, b,
                                                  (short)0, c, false, false);
}

__device__ __forceinline__ float sigmoidf_(float x) {
    return 1.0f / (1.0f + __expf(-x));
}

#if __has_builtin(__builtin_amdgcn_sched_group_barrier)
#define SCHED_GROUP(mask, size, id) \
    __builtin_amdgcn_sched_group_barrier((mask), (size), (id))
#else
#define SCHED_GROUP(mask, size, id)
#endif

// ---------------------------------------------------------------------------
// Kernel 1: f32 -> f16 conversion (weights)
// ---------------------------------------------------------------------------
__global__ void cvt_f32_to_f16(const float* __restrict__ src,
                               _Float16* __restrict__ dst, int n) {
    for (int i = blockIdx.x * blockDim.x + threadIdx.x; i < n;
         i += gridDim.x * blockDim.x)
        dst[i] = (_Float16)src[i];
}

// ---------------------------------------------------------------------------
// Kernel 2: embedding gather -> x[T,B,E] in f16
// ---------------------------------------------------------------------------
__global__ void gather_x(const int* __restrict__ sentence,   // [B,T]
                         const float* __restrict__ emb,      // [V,E]
                         _Float16* __restrict__ xh) {        // [T,B,E]
    const int n = TB * E_;
    for (int i = blockIdx.x * blockDim.x + threadIdx.x; i < n;
         i += gridDim.x * blockDim.x) {
        int e  = i & (E_ - 1);
        int tb = i >> 8;          // E_=256
        int t  = tb >> 5;         // B_=32
        int b  = tb & 31;
        int tok = sentence[b * T_ + t];
        xh[i] = (_Float16)emb[(size_t)tok * E_ + e];
    }
}

// ---------------------------------------------------------------------------
// Kernel 3: xw[d] = x @ W_ih[d]^T + bias[d]
// A-panel (64 rows x 256 K, 32KB) staged in LDS by a single TDM DMA
// (tensor_load_to_lds) with per-row padding for bank-conflict-free frag reads.
// OUTPUT IS SWIZZLED into WMMA C-fragment layout:
//   xw[(((t*256 + mt*128 + n_tile)*32 + lane)*8 + i]
// grid = (2048/128, 8192/64, 2 dirs), block = 256 (8 waves)
// ---------------------------------------------------------------------------
#define AP 264   // LDS A-panel row stride in f16 (512B data + 16B pad)

__global__ void input_gemm(const _Float16* __restrict__ xh,
                           const _Float16* __restrict__ wih_f,
                           const _Float16* __restrict__ wih_b,
                           const float* __restrict__ bias_f,
                           const float* __restrict__ bias_b,
                           _Float16* __restrict__ xw_f,
                           _Float16* __restrict__ xw_b) {
    __shared__ _Float16 sA[64 * AP];    // 33,792 B

    const int dir = blockIdx.z;
    const _Float16* __restrict__ wih  = dir ? wih_b : wih_f;
    const float*    __restrict__ bias = dir ? bias_b : bias_f;
    _Float16*       __restrict__ xw   = dir ? xw_b : xw_f;

    const int wave = threadIdx.x >> 5;
    const int lane = threadIdx.x & 31;
    const int lo = lane & 15, hi = lane >> 4;

    const int m0 = blockIdx.y * 64 + (wave >> 1) * 16;   // row base (T*B dim)
    const int n0 = blockIdx.x * 128 + (wave & 1) * 64;   // col base (4H dim)

#if __has_builtin(__builtin_amdgcn_tensor_load_to_lds) && \
    __has_builtin(__builtin_amdgcn_s_wait_tensorcnt)
    // ---- TDM: DMA the 64x256 f16 A-panel into LDS, 16B pad per 512B row ----
    if (threadIdx.x < 32) {
        const unsigned long long ga =
            (unsigned long long)(const void*)(xh + (size_t)blockIdx.y * 64 * E_);
        const unsigned lds_off =
            (unsigned)(unsigned long long)(const void*)sA;   // LDS aperture low32
        u32x4 g0;
        g0.x = 1u;                                  // count=1, user mode
        g0.y = lds_off;                             // lds_addr (bytes)
        g0.z = (unsigned)(ga & 0xFFFFFFFFull);      // global_addr[31:0]
        g0.w = (unsigned)((ga >> 32) & 0x01FFFFFFull) | (2u << 30); // type=2
        i32x8 g1;
        g1.s0 = (1 << 16)      // data_size = 2 bytes
              | (1 << 20)      // pad_enable
              | (6 << 22)      // pad_interval: 128 DWORDs (512B)
              | (3 << 25);     // pad_amount: 4 DWORDs (16B)
        g1.s1 = (256 << 16);   // tensor_dim0[15:0] = 256 (elements)
        g1.s2 = (64 << 16);    // tensor_dim0 hi = 0 | tensor_dim1[15:0] = 64
        g1.s3 = (256 << 16);   // tensor_dim1 hi = 0 | tile_dim0 = 256
        g1.s4 = 64;            // tile_dim1 = 64, tile_dim2 = 0
        g1.s5 = 256;           // tensor_dim0_stride lo32 = 256 (elements)
        g1.s6 = 0;
        g1.s7 = 0;
        i32x4 gz = {0, 0, 0, 0};
#if __clang_major__ >= 23
        i32x8 gz8 = {0, 0, 0, 0, 0, 0, 0, 0};
        __builtin_amdgcn_tensor_load_to_lds(g0, g1, gz, gz, gz8, 0);
#else
        __builtin_amdgcn_tensor_load_to_lds(g0, g1, gz, gz, 0);
#endif
        __builtin_amdgcn_s_wait_tensorcnt(0);
    }
#else
    // fallback: cooperative staged copy with the same padded layout
    for (int i = threadIdx.x; i < 64 * 256; i += 256) {
        int r = i >> 8, c = i & 255;
        sA[r * AP + c] = xh[(size_t)(blockIdx.y * 64 + r) * E_ + c];
    }
#endif
    __syncthreads();

    const _Float16* aBase = &sA[((wave >> 1) * 16 + lo) * AP + hi * 8];
    const _Float16* bBase[4];
#pragma unroll
    for (int nt = 0; nt < 4; ++nt)
        bBase[nt] = wih + (size_t)(n0 + nt * 16 + lo) * E_ + hi * 8;

    v8f acc[4] = {};

    // software-pipelined K loop (8 iters, 2x unrolled ping-pong B frags)
    Frag16 a0, a1, b0[4], b1[4];
    a0.f4[0] = *(const float4*)(aBase);
    a0.f4[1] = *(const float4*)(aBase + 16);
#pragma unroll
    for (int nt = 0; nt < 4; ++nt) {
        b0[nt].f4[0] = *(const float4*)(bBase[nt]);
        b0[nt].f4[1] = *(const float4*)(bBase[nt] + 16);
    }
#pragma unroll
    for (int kk = 0; kk < 8; kk += 2) {
        const int k1 = kk * 32 + 32;
        a1.f4[0] = *(const float4*)(aBase + k1);
        a1.f4[1] = *(const float4*)(aBase + k1 + 16);
#pragma unroll
        for (int nt = 0; nt < 4; ++nt) {
            b1[nt].f4[0] = *(const float4*)(bBase[nt] + k1);
            b1[nt].f4[1] = *(const float4*)(bBase[nt] + k1 + 16);
        }
#pragma unroll
        for (int nt = 0; nt < 4; ++nt)
            acc[nt] = wmma_f16(a0.v, b0[nt].v, acc[nt]);
        if (kk + 2 < 8) {
            const int k2 = kk * 32 + 64;
            a0.f4[0] = *(const float4*)(aBase + k2);
            a0.f4[1] = *(const float4*)(aBase + k2 + 16);
#pragma unroll
            for (int nt = 0; nt < 4; ++nt) {
                b0[nt].f4[0] = *(const float4*)(bBase[nt] + k2);
                b0[nt].f4[1] = *(const float4*)(bBase[nt] + k2 + 16);
            }
        }
#pragma unroll
        for (int nt = 0; nt < 4; ++nt)
            acc[nt] = wmma_f16(a1.v, b1[nt].v, acc[nt]);
        // interleave hint: 2 WMMA : 2 VMEM-read : 1 DS-read
#pragma unroll
        for (int r = 0; r < 4; ++r) {
            SCHED_GROUP(0x008, 2, 0);   // MFMA/WMMA
            SCHED_GROUP(0x020, 2, 0);   // VMEM read
            SCHED_GROUP(0x100, 1, 0);   // DS read
        }
    }

    // epilogue: bias + f16 convert + swizzled b128 store
    const int mtile_g = m0 >> 4;          // global 16-row tile index
    const int t  = mtile_g >> 1;
    const int mt = mtile_g & 1;
#pragma unroll
    for (int nt = 0; nt < 4; ++nt) {
        const int col    = n0 + nt * 16 + lo;
        const int n_tile = (n0 >> 4) + nt;
        const float bv = bias[col];
        Half8 h8;
#pragma unroll
        for (int i = 0; i < 8; ++i)
            h8.h[i] = (_Float16)(acc[nt][i] + bv);
        *(float4*)(xw +
            (((size_t)t * 256 + mt * 128 + n_tile) * 32 + lane) * 8) = h8.f4;
    }
}

// ---------------------------------------------------------------------------
// Kernel 4: persistent LSTM scan, one workgroup per direction.
// block = 1024 threads = 32 waves. Wave w owns hidden units [16w,16w+16)
// across all 4 gates + both M tiles -> gate update fully in registers,
// c-state lives in VGPRs for the whole scan.
// ---------------------------------------------------------------------------
#define HP 520   // padded LDS row stride (f16 elements, 16B-aligned rows)

__global__ void __launch_bounds__(1024)
lstm_scan(const _Float16* __restrict__ xw_f,   // swizzled [T,256,32,8]
          const _Float16* __restrict__ xw_b,
          const _Float16* __restrict__ whh_f,  // [4H,H] f16
          const _Float16* __restrict__ whh_b,
          const float* __restrict__ h0,        // [2,B,H]
          const float* __restrict__ c0,        // [2,B,H]
          _Float16* __restrict__ hs_f,         // [T,B,H]
          _Float16* __restrict__ hs_b) {
    __shared__ _Float16 h_lds[B_ * HP];

    const int dir = blockIdx.x;
    const _Float16* __restrict__ xw  = dir ? xw_b : xw_f;
    const _Float16* __restrict__ whh = dir ? whh_b : whh_f;
    _Float16*       __restrict__ hs  = dir ? hs_b : hs_f;

    const int tid  = threadIdx.x;
    const int wave = tid >> 5;
    const int lane = tid & 31;
    const int lo = lane & 15, hi = lane >> 4;
    const int j  = wave * 16 + lo;       // hidden unit owned by this lane

    // per-lane invariant W_hh row pointers (one per gate)
    const _Float16* bBase[4];
#pragma unroll
    for (int g = 0; g < 4; ++g)
        bBase[g] = whh + (size_t)(g * H_ + wave * 16 + lo) * H_ + hi * 8;

    // init h in LDS from h0[dir]
    for (int i = tid; i < B_ * H_; i += 1024) {
        int b = i >> 9, jj = i & (H_ - 1);
        h_lds[b * HP + jj] = (_Float16)h0[(size_t)dir * B_ * H_ + i];
    }
    // init c-state in registers: creg[mt][i] = c0[dir][b][j]
    float creg[2][8];
#pragma unroll
    for (int mt = 0; mt < 2; ++mt)
#pragma unroll
        for (int i = 0; i < 8; ++i) {
            int b = mt * 16 + hi * 8 + i;
            creg[mt][i] = c0[(size_t)dir * B_ * H_ + b * H_ + j];
        }
    __syncthreads();

    for (int s = 0; s < T_; ++s) {
        const int tx = dir ? (T_ - 1 - s) : s;

        // prefetch next step's xw C-fragments (global_prefetch_b8)
        if (s + 1 < T_) {
            const int txn = dir ? (T_ - 2 - s) : (s + 1);
#pragma unroll
            for (int mt = 0; mt < 2; ++mt)
#pragma unroll
                for (int g = 0; g < 4; ++g)
                    __builtin_prefetch(xw +
                        (((size_t)txn * 256 + mt * 128 + g * 32 + wave) * 32 +
                         lane) * 8, 0, 1);
        }

        // ---- accumulator init: z = xw[tx] (swizzled, one b128 per tile) ----
        v8f acc[2][4];
#pragma unroll
        for (int mt = 0; mt < 2; ++mt)
#pragma unroll
            for (int g = 0; g < 4; ++g) {
                Half8 h8;
                h8.f4 = *(const float4*)(xw +
                    (((size_t)tx * 256 + mt * 128 + g * 32 + wave) * 32 +
                     lane) * 8);
#pragma unroll
                for (int i = 0; i < 8; ++i)
                    acc[mt][g][i] = (float)h8.h[i];
            }

        // ---- z += h_{s-1} @ Whh^T  (M=32, N=4x16 gate-strided, K=512) ----
        // 16 K-iters, 2x unrolled ping-pong B frags + sched_group_barrier
        // interleave so WMMAs overlap the next iteration's L2 weight loads.
        Frag16 b0[4], b1[4];
#pragma unroll
        for (int g = 0; g < 4; ++g) {
            b0[g].f4[0] = *(const float4*)(bBase[g]);
            b0[g].f4[1] = *(const float4*)(bBase[g] + 16);
        }
        for (int kk = 0; kk < 16; kk += 2) {
            const int k0 = kk * 32;
            // load B for iteration kk+1
#pragma unroll
            for (int g = 0; g < 4; ++g) {
                b1[g].f4[0] = *(const float4*)(bBase[g] + k0 + 32);
                b1[g].f4[1] = *(const float4*)(bBase[g] + k0 + 48);
            }
            {
                Frag16 a0, a1;
                const _Float16* p0 = &h_lds[lo * HP + k0 + hi * 8];
                const _Float16* p1 = &h_lds[(16 + lo) * HP + k0 + hi * 8];
                a0.f4[0] = *(const float4*)(p0);
                a0.f4[1] = *(const float4*)(p0 + 16);
                a1.f4[0] = *(const float4*)(p1);
                a1.f4[1] = *(const float4*)(p1 + 16);
#pragma unroll
                for (int g = 0; g < 4; ++g) {
                    acc[0][g] = wmma_f16(a0.v, b0[g].v, acc[0][g]);
                    acc[1][g] = wmma_f16(a1.v, b0[g].v, acc[1][g]);
                }
            }
            // load B for iteration kk+2
            if (kk + 2 < 16) {
#pragma unroll
                for (int g = 0; g < 4; ++g) {
                    b0[g].f4[0] = *(const float4*)(bBase[g] + k0 + 64);
                    b0[g].f4[1] = *(const float4*)(bBase[g] + k0 + 80);
                }
            }
            {
                Frag16 a0, a1;
                const _Float16* p0 = &h_lds[lo * HP + k0 + 32 + hi * 8];
                const _Float16* p1 = &h_lds[(16 + lo) * HP + k0 + 32 + hi * 8];
                a0.f4[0] = *(const float4*)(p0);
                a0.f4[1] = *(const float4*)(p0 + 16);
                a1.f4[0] = *(const float4*)(p1);
                a1.f4[1] = *(const float4*)(p1 + 16);
#pragma unroll
                for (int g = 0; g < 4; ++g) {
                    acc[0][g] = wmma_f16(a0.v, b1[g].v, acc[0][g]);
                    acc[1][g] = wmma_f16(a1.v, b1[g].v, acc[1][g]);
                }
            }
            // interleave hint: 2 WMMA : 1 VMEM-read : 1 DS-read, x8
#pragma unroll
            for (int r = 0; r < 8; ++r) {
                SCHED_GROUP(0x008, 2, 0);   // MFMA/WMMA
                SCHED_GROUP(0x020, 1, 0);   // VMEM read
                SCHED_GROUP(0x100, 1, 0);   // DS read
            }
        }

        // all reads of old h finished; wait for everyone
        __syncthreads();

        // ---- gate nonlinearities + c/h update, fully in registers ----
#pragma unroll
        for (int mt = 0; mt < 2; ++mt) {
#pragma unroll
            for (int i = 0; i < 8; ++i) {
                int b = mt * 16 + hi * 8 + i;
                float ig = sigmoidf_(acc[mt][0][i]);
                float fg = sigmoidf_(acc[mt][1][i]);
                float gg = tanhf(acc[mt][2][i]);
                float og = sigmoidf_(acc[mt][3][i]);
                float c  = fg * creg[mt][i] + ig * gg;
                creg[mt][i] = c;
                float h = og * tanhf(c);
                h_lds[b * HP + j] = (_Float16)h;
            }
        }
        __syncthreads();

        // ---- vectorized h writeback: whole WG copies h_lds -> hs[tx] ----
        {
            const int bb  = tid >> 5;
            const int off = (tid & 31) * 16;      // 16 f16 = 32B per thread
            const float4* src = (const float4*)&h_lds[bb * HP + off];
            float4* dst =
                (float4*)&hs[((size_t)tx * B_ + bb) * H_ + off];
            dst[0] = src[0];
            dst[1] = src[1];
        }
    }
}

// ---------------------------------------------------------------------------
// Kernel 5: feats[b,t,:] = [hf ; hb] @ W_out^T + b_out   (M=8192,N=32,K=1024)
// block = 256 (8 waves), wave = one 16-row M tile x 2 N tiles, pipelined K.
// ---------------------------------------------------------------------------
__global__ void feats_gemm(const _Float16* __restrict__ hf,   // [T,B,H]
                           const _Float16* __restrict__ hb,   // [T,B,H]
                           const _Float16* __restrict__ wout, // [K,2H] f16
                           const float* __restrict__ b_out,   // [K]
                           float* __restrict__ feats) {       // [B,T,K]
    const int wave = threadIdx.x >> 5;
    const int lane = threadIdx.x & 31;
    const int lo = lane & 15, hi = lane >> 4;
    const int m_tile = blockIdx.x * 8 + wave;
    const int r = m_tile * 16 + lo;           // flat row = b*T + t
    const int b = r >> 8, t = r & (T_ - 1);

    const _Float16* aF = hf + ((size_t)t * B_ + b) * H_ + hi * 8;
    const _Float16* aB = hb + ((size_t)t * B_ + b) * H_ + hi * 8;
    const _Float16* bBase[2];
#pragma unroll
    for (int nt = 0; nt < 2; ++nt)
        bBase[nt] = wout + (size_t)(nt * 16 + lo) * (2 * H_) + hi * 8;

    v8f acc[2] = {};

    Frag16 a0, a1, b0[2], b1[2];
    a0.f4[0] = *(const float4*)(aF);
    a0.f4[1] = *(const float4*)(aF + 16);
#pragma unroll
    for (int nt = 0; nt < 2; ++nt) {
        b0[nt].f4[0] = *(const float4*)(bBase[nt]);
        b0[nt].f4[1] = *(const float4*)(bBase[nt] + 16);
    }
    for (int kk = 0; kk < 32; kk += 2) {
        const int k1 = kk * 32 + 32;
        {
            const _Float16* src = (k1 < H_) ? aF : aB;
            const int kkk = k1 & (H_ - 1);
            a1.f4[0] = *(const float4*)(src + kkk);
            a1.f4[1] = *(const float4*)(src + kkk + 16);
        }
#pragma unroll
        for (int nt = 0; nt < 2; ++nt) {
            b1[nt].f4[0] = *(const float4*)(bBase[nt] + k1);
            b1[nt].f4[1] = *(const float4*)(bBase[nt] + k1 + 16);
        }
#pragma unroll
        for (int nt = 0; nt < 2; ++nt)
            acc[nt] = wmma_f16(a0.v, b0[nt].v, acc[nt]);
        if (kk + 2 < 32) {
            const int k2 = kk * 32 + 64;
            const _Float16* src = (k2 < H_) ? aF : aB;
            const int kkk = k2 & (H_ - 1);
            a0.f4[0] = *(const float4*)(src + kkk);
            a0.f4[1] = *(const float4*)(src + kkk + 16);
#pragma unroll
            for (int nt = 0; nt < 2; ++nt) {
                b0[nt].f4[0] = *(const float4*)(bBase[nt] + k2);
                b0[nt].f4[1] = *(const float4*)(bBase[nt] + k2 + 16);
            }
        }
#pragma unroll
        for (int nt = 0; nt < 2; ++nt)
            acc[nt] = wmma_f16(a1.v, b1[nt].v, acc[nt]);
#pragma unroll
        for (int r2 = 0; r2 < 4; ++r2) {
            SCHED_GROUP(0x008, 1, 0);   // WMMA
            SCHED_GROUP(0x020, 2, 0);   // VMEM read
        }
    }
#pragma unroll
    for (int nt = 0; nt < 2; ++nt) {
        const int col = nt * 16 + lo;
        const float bv = b_out[col];
#pragma unroll
        for (int i = 0; i < 8; ++i) {
            const int row = m_tile * 16 + hi * 8 + i;  // = b*T + t
            feats[(size_t)row * K_ + col] = acc[nt][i] + bv;
        }
    }
}

// ---------------------------------------------------------------------------
// Kernel 6: CRF Viterbi decode. 1 block, 1024 threads = (batch b, tag nx).
// ---------------------------------------------------------------------------
__global__ void __launch_bounds__(1024)
viterbi(const float* __restrict__ feats,       // [B,T,K]
        const float* __restrict__ trans_in,    // [K,K] trans[next][prev]
        unsigned char* __restrict__ bps,       // [T,B,K]
        float* __restrict__ out) {             // [B] score ++ [B,T] path
    __shared__ float vbuf[2][B_][K_];
    __shared__ float tr[K_][K_];

    const int b  = threadIdx.x >> 5;
    const int nx = threadIdx.x & 31;

    tr[b][nx]      = trans_in[b * K_ + nx];    // (b reused as "next" here)
    vbuf[0][b][nx] = (nx == START_) ? 0.0f : NEG_;
    __syncthreads();

    int cur = 0;
    for (int t = 0; t < T_; ++t) {
        float best = -3.0e38f;
        int   bp   = 0;
#pragma unroll
        for (int prev = 0; prev < K_; ++prev) {
            float s = vbuf[cur][b][prev] + tr[nx][prev];
            if (s > best) { best = s; bp = prev; }
        }
        vbuf[cur ^ 1][b][nx] = best + feats[((size_t)b * T_ + t) * K_ + nx];
        bps[((size_t)t * B_ + b) * K_ + nx] = (unsigned char)bp;
        cur ^= 1;
        __syncthreads();
    }

    // terminal + argmax over tags (lane reduction, wave32)
    float bv = vbuf[cur][b][nx] + tr[STOP_][nx];
    int   bi = nx;
    for (int off = 16; off > 0; off >>= 1) {
        float ov = __shfl_down(bv, off);
        int   oi = __shfl_down(bi, off);
        if (ov > bv || (ov == bv && oi < bi)) { bv = ov; bi = oi; }
    }
    __threadfence_block();
    __syncthreads();

    if (nx == 0) {
        out[b] = bv;
        int tag = bi;
        for (int t = T_ - 1; t >= 0; --t) {
            out[B_ + b * T_ + t] = (float)tag;
            tag = (int)bps[((size_t)t * B_ + b) * K_ + tag];
        }
    }
}

// ---------------------------------------------------------------------------
// Host launcher
// ---------------------------------------------------------------------------
extern "C" void kernel_launch(void* const* d_in, const int* in_sizes, int n_in,
                              void* d_out, int out_size, void* d_ws, size_t ws_size,
                              hipStream_t stream) {
    const int*   sentence = (const int*)  d_in[0];
    const float* emb      = (const float*)d_in[1];
    const float* W_ih_f   = (const float*)d_in[2];
    const float* W_hh_f   = (const float*)d_in[3];
    const float* b_f      = (const float*)d_in[4];
    const float* W_ih_b   = (const float*)d_in[5];
    const float* W_hh_b   = (const float*)d_in[6];
    const float* b_b      = (const float*)d_in[7];
    const float* W_out    = (const float*)d_in[8];
    const float* b_out    = (const float*)d_in[9];
    const float* trans    = (const float*)d_in[10];
    const float* h0       = (const float*)d_in[11];
    const float* c0       = (const float*)d_in[12];
    float* out = (float*)d_out;

    // ---- workspace carving (all offsets 256B aligned) ----
    char* p = (char*)d_ws;
    auto carve = [&](size_t bytes) {
        char* r = p;
        p += (bytes + 255) & ~(size_t)255;
        return r;
    };
    _Float16* xh     = (_Float16*)carve((size_t)TB * E_ * 2);        // 4 MB
    _Float16* wih16f = (_Float16*)carve((size_t)G4H * E_ * 2);       // 1 MB
    _Float16* wih16b = (_Float16*)carve((size_t)G4H * E_ * 2);       // 1 MB
    _Float16* whh16f = (_Float16*)carve((size_t)G4H * H_ * 2);       // 2 MB
    _Float16* whh16b = (_Float16*)carve((size_t)G4H * H_ * 2);       // 2 MB
    _Float16* wout16 = (_Float16*)carve((size_t)K_ * 2 * H_ * 2);    // 64 KB
    _Float16* xw_f   = (_Float16*)carve((size_t)TB * G4H * 2);       // 32 MB
    _Float16* xw_b   = (_Float16*)carve((size_t)TB * G4H * 2);       // 32 MB
    _Float16* hs_f   = (_Float16*)carve((size_t)TB * H_ * 2);        // 8 MB
    _Float16* hs_b   = (_Float16*)carve((size_t)TB * H_ * 2);        // 8 MB
    float*    feats  = (float*)   carve((size_t)TB * K_ * 4);        // 1 MB
    unsigned char* bps = (unsigned char*)carve((size_t)T_ * B_ * K_);// 256 KB

    // 1) weight conversions
    cvt_f32_to_f16<<<512, 256, 0, stream>>>(W_ih_f, wih16f, G4H * E_);
    cvt_f32_to_f16<<<512, 256, 0, stream>>>(W_ih_b, wih16b, G4H * E_);
    cvt_f32_to_f16<<<512, 256, 0, stream>>>(W_hh_f, whh16f, G4H * H_);
    cvt_f32_to_f16<<<512, 256, 0, stream>>>(W_hh_b, whh16b, G4H * H_);
    cvt_f32_to_f16<<<128, 256, 0, stream>>>(W_out,  wout16, K_ * 2 * H_);

    // 2) embedding gather
    gather_x<<<2048, 256, 0, stream>>>(sentence, emb, xh);

    // 3) xw = x @ W_ih^T + b, TDM-staged A panel, swizzled output  (WMMA)
    {
        dim3 grid(G4H / 128, TB / 64, 2);
        input_gemm<<<grid, 256, 0, stream>>>(xh, wih16f, wih16b, b_f, b_b,
                                             xw_f, xw_b);
    }

    // 4) recurrent scan, one persistent WG per direction  (WMMA)
    lstm_scan<<<2, 1024, 0, stream>>>(xw_f, xw_b, whh16f, whh16b,
                                      h0, c0, hs_f, hs_b);

    // 5) output projection  (WMMA)
    feats_gemm<<<TB / (16 * 8), 256, 0, stream>>>(hs_f, hs_b, wout16, b_out,
                                                  feats);

    // 6) Viterbi decode
    viterbi<<<1, 1024, 0, stream>>>(feats, trans, bps, out);
}
